// Darknet_2_28638841930106
// MI455X (gfx1250) — compile-verified, matrix operand order checked by reading, not run on hardware
//
#include <hip/hip_runtime.h>

// CDNA5 / gfx1250 masked conv2d (3x3, pad=1) as implicit GEMM on bf16 WMMA.
// out[f,h,w] = (sum_{c,kh,kw} x[c,h+kh-1,w+kw-1]*wt[f,c,kh,kw]) * mask[h,w] + b[f]
//
// Fast path (needs ~41 MB of d_ws):
//   pack_weights: wt -> a_pack[tap][f][c]  (bf16)
//   pack_x      : x  -> x_t[h][w][c]       (bf16, channel-contiguous)
//   main kernel : 128(F) x 192(full row) block tile, 8 waves of 64x48,
//                 36 WMMAs per sync pair; staging pipelined via
//                 global_load_async_to_lds_b128 (if available) or register
//                 double-buffering.

typedef __attribute__((ext_vector_type(16))) __bf16 v16bf;
typedef __attribute__((ext_vector_type(8)))  float  v8f;

#define H_   192
#define W_   192
#define C_   512
#define F_   512
#define HW_  (H_ * W_)
#define CKK_ (C_ * 9)
#define FC_  (F_ * C_)

#define LDA 40    // padded LDS row stride in bf16 elems (80B = 20 banks, 16B aligned)

#define BM    128  // filter tile
#define BN    192  // full output row
#define BK    32   // channel chunk = bf16 WMMA K
#define BROWS 194  // staged strip: wsrc in [-1, 192]; rows 0 and 193 are zero halo

#if defined(__gfx1250__) && \
    __has_builtin(__builtin_amdgcn_global_load_async_to_lds_b128) && \
    __has_builtin(__builtin_amdgcn_s_wait_asynccnt)
#define USE_ASYNC 1
#else
#define USE_ASYNC 0
#endif

// fp32 -> bf16, round-to-nearest-even
__device__ __forceinline__ __bf16 f2bf(float f) {
  union { float f; unsigned u; } v; v.f = f;
  unsigned r = (v.u + 0x7FFFu + ((v.u >> 16) & 1u)) >> 16;
  union { unsigned short s; __bf16 b; } o; o.s = (unsigned short)r;
  return o.b;
}

// Load one 16x32 bf16 A/B fragment from an LDS tile stored [row][k], k-major.
// ISA layout (CDNA5 ISA, 16-bit A 16x32):
//   lanes 0-15 : row = lane,    K = {0..7, 16..23}
//   lanes 16-31: row = lane-16, K = {8..15, 24..31}
__device__ __forceinline__ v16bf load_frag(const __bf16* tile, int row_base, int lane) {
  const __bf16* p = tile + (row_base + (lane & 15)) * LDA;
  const int lo = (lane >> 4) & 1;
  union { float4 v; __bf16 h[8]; } c0, c1;
  c0.v = ((const float4*)p)[lo];
  c1.v = ((const float4*)p)[lo + 2];
  v16bf f;
#pragma unroll
  for (int i = 0; i < 8; ++i) { f[i] = c0.h[i]; f[8 + i] = c1.h[i]; }
  return f;
}

// 3 kw planes x (4x3) tiles = 36 WMMAs
__device__ __forceinline__ void compute_tiles(const __bf16* la, const __bf16* lb,
                                              v8f (&acc)[4][3], int wm, int wn, int lane) {
#pragma unroll
  for (int kw = 0; kw < 3; ++kw) {
    const __bf16* ap = la + kw * BM * LDA;
    v16bf af[4], bfm[3];
#pragma unroll
    for (int tm = 0; tm < 4; ++tm) af[tm] = load_frag(ap, wm * 64 + tm * 16, lane);
#pragma unroll
    for (int tn = 0; tn < 3; ++tn) bfm[tn] = load_frag(lb, wn * 48 + tn * 16 + kw, lane);
#pragma unroll
    for (int tm = 0; tm < 4; ++tm)
#pragma unroll
      for (int tn = 0; tn < 3; ++tn)
        acc[tm][tn] = __builtin_amdgcn_wmma_f32_16x16x32_bf16(
            false, af[tm], false, bfm[tn], (short)0, acc[tm][tn], false, false);
  }
}

// D layout: VGPR r holds M = r (+8 for lanes 16-31), N = lane&15
__device__ __forceinline__ void epilogue(const v8f (&acc)[4][3], int wm, int wn, int lane,
                                         int h, int f0, const int* mask,
                                         const float* bias, float* out) {
  const int lhalf = (lane >> 4) & 1;
  const int lmod  = lane & 15;
#pragma unroll
  for (int tn = 0; tn < 3; ++tn) {
    const int   np = wn * 48 + tn * 16 + lmod;
    const float mv = mask[h * W_ + np] ? 1.0f : 0.0f;
#pragma unroll
    for (int tm = 0; tm < 4; ++tm) {
      const int fb = f0 + wm * 64 + tm * 16 + lhalf * 8;
#pragma unroll
      for (int r = 0; r < 8; ++r)
        out[(size_t)(fb + r) * HW_ + (size_t)h * W_ + np] =
            acc[tm][tn][r] * mv + bias[fb + r];
    }
  }
}

// ---------------- pack kernels ----------------

__global__ __launch_bounds__(256)
void pack_weights(const float* __restrict__ wt, __bf16* __restrict__ a_pack) {
  const int e = blockIdx.x * 256 + threadIdx.x;
  const int f = e >> 9;
  const int c = e & 511;
  const float* s = wt + (size_t)f * CKK_ + (size_t)c * 9;
#pragma unroll
  for (int t = 0; t < 9; ++t)
    a_pack[(size_t)t * FC_ + (size_t)f * C_ + c] = f2bf(s[t]);
}

__global__ __launch_bounds__(256)
void pack_x(const float* __restrict__ x, __bf16* __restrict__ x_t) {
  const int hw0 = blockIdx.x * 8;
#pragma unroll
  for (int it = 0; it < 2; ++it) {
    const int c = threadIdx.x + it * 256;
    const float4 v0 = *(const float4*)(x + (size_t)c * HW_ + hw0);
    const float4 v1 = *(const float4*)(x + (size_t)c * HW_ + hw0 + 4);
    const float vv[8] = {v0.x, v0.y, v0.z, v0.w, v1.x, v1.y, v1.z, v1.w};
#pragma unroll
    for (int j = 0; j < 8; ++j)
      x_t[(size_t)(hw0 + j) * C_ + c] = f2bf(vv[j]);
  }
}

// ---------------- main kernel ----------------

#if USE_ASYNC

// Builtin signature (from hipcc diagnostic): arg0 = v4i* in AS1 (global src,
// printed as "__device__"), arg1 = LDS dest, arg2/arg3 = offset / cpol.
typedef int v4i __attribute__((vector_size(16)));
typedef __attribute__((address_space(1))) v4i gv4i;
typedef __attribute__((address_space(3))) v4i lv4i;

__device__ __forceinline__ void async_cp16(__bf16* lds_dst, const __bf16* gsrc) {
  __builtin_amdgcn_global_load_async_to_lds_b128((gv4i*)gsrc, (lv4i*)lds_dst, 0, 0);
}

// Issue one chunk's staging: A = 6 units/thread, B = 3 units/thread (uniform).
__device__ __forceinline__ void stage_async(int kh, bool hvalid, int hsrc, int f0, int c0,
                                            __bf16* la, __bf16* lb,
                                            const __bf16* a_pack, const __bf16* x_t,
                                            int tid) {
#pragma unroll
  for (int j = 0; j < 6; ++j) {
    const int u = tid + j * 256;           // 3 planes * 128 rows * 4 = 1536 units
    const int p = u >> 9, rr = (u >> 2) & 127, q = u & 3;
    async_cp16(&la[(p * BM + rr) * LDA + q * 8],
               a_pack + (size_t)(kh * 3 + p) * FC_ + (size_t)(f0 + rr) * C_ + c0 + q * 8);
  }
  if (hvalid) {
#pragma unroll
    for (int j = 0; j < 3; ++j) {
      const int u = tid + j * 256;         // rows 1..192 * 4 = 768 units
      const int n = 1 + (u >> 2), q = u & 3;
      async_cp16(&lb[n * LDA + q * 8],
                 x_t + ((size_t)hsrc * W_ + (n - 1)) * C_ + c0 + q * 8);
    }
  }
}

__global__ __launch_bounds__(256)
void conv_wmma_main(const __bf16* __restrict__ a_pack, const __bf16* __restrict__ x_t,
                    const float* __restrict__ bias, const int* __restrict__ mask,
                    float* __restrict__ out) {
  __shared__ __align__(16) __bf16 lds_a[2][3 * BM * LDA];
  __shared__ __align__(16) __bf16 lds_b[2][BROWS * LDA];

  const int tid = threadIdx.x, lane = tid & 31, wave = tid >> 5;
  const int wm = wave & 1, wn = wave >> 1;
  const int h  = blockIdx.x;
  const int f0 = blockIdx.y * BM;

  v8f acc[4][3];
#pragma unroll
  for (int tm = 0; tm < 4; ++tm)
#pragma unroll
    for (int tn = 0; tn < 3; ++tn)
#pragma unroll
      for (int r = 0; r < 8; ++r) acc[tm][tn][r] = 0.0f;

  // zero the halo rows (0, 193) of both B buffers once; never async-written
  if (tid < 16) {
    const int b = tid >> 3, rsel = (tid >> 2) & 1, q = tid & 3;
    const int n = rsel ? (BROWS - 1) : 0;
    *(uint4*)(&lds_b[b][n * LDA + q * 8]) = make_uint4(0u, 0u, 0u, 0u);
  }
  __syncthreads();

  for (int kh = 0; kh < 3; ++kh) {
    const int  hsrc   = h + kh - 1;
    const bool hvalid = (0 <= hsrc) && (hsrc < H_);
    if (!hvalid) {  // block-uniform: zero both B strips, skip B async this kh
#pragma unroll
      for (int i = 0; i < 7; ++i) {
        const int u = tid + i * 256;
        if (u < 2 * BROWS * 4) {
          const int b = (u >= BROWS * 4) ? 1 : 0;
          const int v = u - b * BROWS * 4;
          *(uint4*)(&lds_b[b][(v >> 2) * LDA + (v & 3) * 8]) = make_uint4(0u, 0u, 0u, 0u);
        }
      }
    }
    __syncthreads();
    stage_async(kh, hvalid, hsrc, f0, 0, lds_a[0], lds_b[0], a_pack, x_t, tid);
    for (int it = 0; it < 16; ++it) {
      const int cur = it & 1;
      const int c0  = it * BK;
      if (it + 1 < 16) {
        stage_async(kh, hvalid, hsrc, f0, c0 + BK,
                    lds_a[1 - cur], lds_b[1 - cur], a_pack, x_t, tid);
        if (hvalid) __builtin_amdgcn_s_wait_asynccnt(9);
        else        __builtin_amdgcn_s_wait_asynccnt(6);
      } else {
        __builtin_amdgcn_s_wait_asynccnt(0);
      }
      __syncthreads();
      compute_tiles(lds_a[cur], lds_b[cur], acc, wm, wn, lane);
      __syncthreads();
    }
  }
  epilogue(acc, wm, wn, lane, h, f0, mask, bias, out);
}

#else  // !USE_ASYNC: register double-buffered pipeline, single LDS buffers

__global__ __launch_bounds__(256)
void conv_wmma_main(const __bf16* __restrict__ a_pack, const __bf16* __restrict__ x_t,
                    const float* __restrict__ bias, const int* __restrict__ mask,
                    float* __restrict__ out) {
  __shared__ __align__(16) __bf16 lds_a[3 * BM * LDA];
  __shared__ __align__(16) __bf16 lds_b[BROWS * LDA];

  const int tid = threadIdx.x, lane = tid & 31, wave = tid >> 5;
  const int wm = wave & 1, wn = wave >> 1;
  const int h  = blockIdx.x;
  const int f0 = blockIdx.y * BM;

  v8f acc[4][3];
#pragma unroll
  for (int tm = 0; tm < 4; ++tm)
#pragma unroll
    for (int tn = 0; tn < 3; ++tn)
#pragma unroll
      for (int r = 0; r < 8; ++r) acc[tm][tn][r] = 0.0f;

  uint4 ra[6], rb[4];
  auto load_regs = [&](int it) {
    const int kh = it >> 4, c0 = (it & 15) * BK;
    const int hsrc = h + kh - 1;
    const bool hvalid = (0 <= hsrc) && (hsrc < H_);
#pragma unroll
    for (int j = 0; j < 6; ++j) {
      const int u = tid + j * 256;       // 3 planes * 128 rows * 4 units
      const int p = u >> 9, rr = (u >> 2) & 127, q = u & 3;
      ra[j] = *(const uint4*)(a_pack + (size_t)(kh * 3 + p) * FC_ +
                              (size_t)(f0 + rr) * C_ + c0 + q * 8);
    }
#pragma unroll
    for (int j = 0; j < 4; ++j) {
      const int u = tid + j * 256;       // 194 rows * 4 = 776 units
      rb[j] = make_uint4(0u, 0u, 0u, 0u);
      if (u < BROWS * 4) {
        const int n = u >> 2, q = u & 3;
        const int wsrc = n - 1;
        if (hvalid && 0 <= wsrc && wsrc < W_)
          rb[j] = *(const uint4*)(x_t + ((size_t)hsrc * W_ + wsrc) * C_ + c0 + q * 8);
      }
    }
  };

  load_regs(0);
  for (int it = 0; it < 48; ++it) {
    // regs -> LDS
#pragma unroll
    for (int j = 0; j < 6; ++j) {
      const int u = tid + j * 256;
      const int p = u >> 9, rr = (u >> 2) & 127, q = u & 3;
      *(uint4*)(&lds_a[(p * BM + rr) * LDA + q * 8]) = ra[j];
    }
#pragma unroll
    for (int j = 0; j < 4; ++j) {
      const int u = tid + j * 256;
      if (u < BROWS * 4)
        *(uint4*)(&lds_b[(u >> 2) * LDA + (u & 3) * 8]) = rb[j];
    }
    __syncthreads();
    if (it + 1 < 48) load_regs(it + 1);  // overlap next chunk's global loads with WMMAs
    compute_tiles(lds_a, lds_b, acc, wm, wn, lane);
    __syncthreads();
  }
  epilogue(acc, wm, wn, lane, h, f0, mask, bias, out);
}

#endif  // USE_ASYNC

// ---------------- fallback (self-contained, no workspace) ----------------

#define FBM 128
#define FBN 64

__global__ __launch_bounds__(256, 2)
void conv_wmma_fallback(const float* __restrict__ x, const float* __restrict__ wt,
                        const float* __restrict__ bias, const int* __restrict__ mask,
                        float* __restrict__ out) {
  __shared__ __align__(16) __bf16 lds_a[FBM * LDA];
  __shared__ __align__(16) __bf16 lds_b[FBN * LDA];
  const int tid = threadIdx.x, lane = tid & 31, wave = tid >> 5;
  const int wm = wave & 3, wn = wave >> 2;
  const int h = blockIdx.x / (W_ / FBN);
  const int w0 = (blockIdx.x % (W_ / FBN)) * FBN;
  const int f0 = blockIdx.y * FBM;
  v8f acc[2][2];
#pragma unroll
  for (int tm = 0; tm < 2; ++tm)
#pragma unroll
    for (int tn = 0; tn < 2; ++tn)
#pragma unroll
      for (int r = 0; r < 8; ++r) acc[tm][tn][r] = 0.0f;
  for (int kh = 0; kh < 3; ++kh) {
    const int hsrc = h + kh - 1;
    const bool hvalid = (0 <= hsrc) && (hsrc < H_);
    for (int kw = 0; kw < 3; ++kw) {
      const int tap = kh * 3 + kw;
      for (int c0 = 0; c0 < C_; c0 += 32) {
#pragma unroll
        for (int i = 0; i < 16; ++i) {
          const int e = tid + i * 256, fl = e >> 5, cl = e & 31;
          lds_a[fl * LDA + cl] =
              f2bf(wt[(size_t)(f0 + fl) * CKK_ + (size_t)(c0 + cl) * 9 + tap]);
        }
#pragma unroll
        for (int i = 0; i < 8; ++i) {
          const int e = tid + i * 256, cl = e >> 6, nl = e & 63;
          const int wsrc = w0 + nl + kw - 1;
          float v = 0.0f;
          if (hvalid && (unsigned)wsrc < (unsigned)W_)
            v = x[(size_t)(c0 + cl) * HW_ + (size_t)hsrc * W_ + wsrc];
          lds_b[nl * LDA + cl] = f2bf(v);
        }
        __syncthreads();
        v16bf af[2], bfm[2];
        af[0] = load_frag(lds_a, wm * 32 + 0, lane);
        af[1] = load_frag(lds_a, wm * 32 + 16, lane);
        bfm[0] = load_frag(lds_b, wn * 32 + 0, lane);
        bfm[1] = load_frag(lds_b, wn * 32 + 16, lane);
#pragma unroll
        for (int tm = 0; tm < 2; ++tm)
#pragma unroll
          for (int tn = 0; tn < 2; ++tn)
            acc[tm][tn] = __builtin_amdgcn_wmma_f32_16x16x32_bf16(
                false, af[tm], false, bfm[tn], (short)0, acc[tm][tn], false, false);
        __syncthreads();
      }
    }
  }
  const int lhalf = (lane >> 4) & 1, lmod = lane & 15;
#pragma unroll
  for (int tn = 0; tn < 2; ++tn) {
    const int np = w0 + wn * 32 + tn * 16 + lmod;
    const float mv = mask[h * W_ + np] ? 1.0f : 0.0f;
#pragma unroll
    for (int tm = 0; tm < 2; ++tm) {
      const int fb = f0 + wm * 32 + tm * 16 + lhalf * 8;
#pragma unroll
      for (int r = 0; r < 8; ++r)
        out[(size_t)(fb + r) * HW_ + (size_t)h * W_ + np] =
            acc[tm][tn][r] * mv + bias[fb + r];
    }
  }
}

// ---------------- launcher ----------------

extern "C" void kernel_launch(void* const* d_in, const int* in_sizes, int n_in,
                              void* d_out, int out_size, void* d_ws, size_t ws_size,
                              hipStream_t stream) {
  (void)in_sizes; (void)n_in; (void)out_size;
  const float* x    = (const float*)d_in[0];
  const float* wt   = (const float*)d_in[1];
  const float* bias = (const float*)d_in[2];
  const int*   mask = (const int*)d_in[3];
  float*       out  = (float*)d_out;

  const size_t A_BYTES = (size_t)9 * FC_ * sizeof(__bf16);
  const size_t X_BYTES = (size_t)HW_ * C_ * sizeof(__bf16);
  if (ws_size >= A_BYTES + X_BYTES) {
    __bf16* a_pack = (__bf16*)d_ws;
    __bf16* x_t    = (__bf16*)((char*)d_ws + A_BYTES);
    pack_weights<<<dim3(FC_ / 256), dim3(256), 0, stream>>>(wt, a_pack);
    pack_x<<<dim3(HW_ / 8), dim3(256), 0, stream>>>(x, x_t);
    conv_wmma_main<<<dim3(H_, F_ / BM), dim3(256), 0, stream>>>(a_pack, x_t, bias, mask, out);
  } else {
    conv_wmma_fallback<<<dim3(H_ * (W_ / FBN), F_ / FBM), dim3(256), 0, stream>>>(
        x, wt, bias, mask, out);
  }
}
